// Head_1262720385518
// MI455X (gfx1250) — matrix operand order repaired and checked by reading, not compiled, not run on hardware
//
#include <hip/hip_runtime.h>

// ---------------------------------------------------------------------------
// Single-head causal attention for MI455X (gfx1250, wave32, WMMA).
//   B=4, T=4096, C=1024, H=64.
//   Stage 0: transpose W{q,k,v} -> f16 [n][k] once (384 KB, L2-resident).
//   Stage 1: q/k/v projections via V_WMMA_F32_16X16X32_F16, no LDS/barriers.
//   Stage 2: flash-attention (online softmax, f32 state).
// Fragment loads are batched ahead of each WMMA group and pinned with
// sched_barrier so the per-WMMA s_wait_loadcnt becomes partial (overlapped)
// instead of a full wait per matrix op.
// ---------------------------------------------------------------------------

typedef __attribute__((ext_vector_type(16))) _Float16     v16h;
typedef __attribute__((ext_vector_type(8)))  float        v8f;
typedef __attribute__((ext_vector_type(4)))  unsigned int u32x4;

union Frag { v16h h; u32x4 u[2]; _Float16 f[16]; };

#define T_SEQ 4096
#define C_EMB 1024
#define HDIM  64

static __device__ inline v8f vzero() {
  v8f z;
  for (int i = 0; i < 8; ++i) z[i] = 0.0f;
  return z;
}

static __device__ inline v8f wmma_f16(v16h a, v16h b, v8f c) {
  // D = A(16x32 f16) * B(32x16 f16) + C(16x16 f32)
  return __builtin_amdgcn_wmma_f32_16x16x32_f16(
      /*neg_a=*/false, a, /*neg_b=*/false, b,
      /*c_mod=*/(short)0, c, /*reuse_a=*/false, /*reuse_b=*/false);
}

static __device__ inline void sched_fence() {
  __builtin_amdgcn_sched_barrier(0);   // keep loads before, WMMAs after
}

// ---------------------------------------------------------------------------
// Kernel 0: wT[m][n][k] = (f16) W_m[k][n]   (one-off, 384 KB output)
// ---------------------------------------------------------------------------
__global__ __launch_bounds__(256) void w_transpose_kernel(
    const float* __restrict__ Wq,
    const float* __restrict__ Wk,
    const float* __restrict__ Wv,
    _Float16* __restrict__ wT)
{
  const int idx = blockIdx.x * 256 + threadIdx.x;   // over 3*1024*64
  const int m   = idx >> 16;
  const int rem = idx & 65535;
  const int k   = rem >> 6;
  const int n   = rem & 63;
  const float* W = (m == 0) ? Wq : ((m == 1) ? Wk : Wv);
  wT[(long)(m * HDIM + n) * C_EMB + k] = (_Float16)W[k * HDIM + n];
}

// ---------------------------------------------------------------------------
// Kernel 1: q = x*Wq, k = x*Wk, v = x*Wv   (f32 -> f16 outputs)
//   q,k stored row-major [b*t][h]; v stored transposed [b][h][t].
// Block: 128 threads = 4 fully independent waves; each wave: 16 rows x 64.
// ---------------------------------------------------------------------------
__global__ __launch_bounds__(128) void qkv_proj_kernel(
    const float* __restrict__ x,
    const _Float16* __restrict__ wT,      // [3][64][1024]
    _Float16* __restrict__ qh,
    _Float16* __restrict__ kh,
    _Float16* __restrict__ vT)
{
  const int tid  = threadIdx.x;
  const int wave = tid >> 5;
  const int lane = tid & 31;
  const int lg   = lane >> 4;   // half-wave group
  const int ln   = lane & 15;
  const long r0  = (long)blockIdx.x * 64 + wave * 16;  // wave's first row

  const float* xrow = x + (r0 + ln) * C_EMB;

  v8f z = vzero();
  v8f acc[3][4];
  #pragma unroll
  for (int m = 0; m < 3; ++m)
    #pragma unroll
    for (int nt = 0; nt < 4; ++nt) acc[m][nt] = z;

  for (int kk = 0; kk < C_EMB; kk += 32) {
    // ---- batch ALL loads for this K-step (A + 12 B fragments) ----
    const float* xp = xrow + kk;
    float4 a0 = *(const float4*)(xp + 8 * lg);
    float4 a1 = *(const float4*)(xp + 8 * lg + 4);
    float4 a2 = *(const float4*)(xp + 16 + 8 * lg);
    float4 a3 = *(const float4*)(xp + 16 + 8 * lg + 4);

    Frag b[12];
    #pragma unroll
    for (int m = 0; m < 3; ++m) {
      #pragma unroll
      for (int nt = 0; nt < 4; ++nt) {
        const _Float16* wp =
            wT + (long)(m * HDIM + nt * 16 + ln) * C_EMB + kk;
        b[m * 4 + nt].u[0] = *(const u32x4*)(wp + 8 * lg);
        b[m * 4 + nt].u[1] = *(const u32x4*)(wp + 16 + 8 * lg);
      }
    }

    // A fragment: element e<8 -> K = e + 8*g ; e>=8 -> K = e + 8 + 8*g
    Frag a;
    a.f[0]  = (_Float16)a0.x; a.f[1]  = (_Float16)a0.y;
    a.f[2]  = (_Float16)a0.z; a.f[3]  = (_Float16)a0.w;
    a.f[4]  = (_Float16)a1.x; a.f[5]  = (_Float16)a1.y;
    a.f[6]  = (_Float16)a1.z; a.f[7]  = (_Float16)a1.w;
    a.f[8]  = (_Float16)a2.x; a.f[9]  = (_Float16)a2.y;
    a.f[10] = (_Float16)a2.z; a.f[11] = (_Float16)a2.w;
    a.f[12] = (_Float16)a3.x; a.f[13] = (_Float16)a3.y;
    a.f[14] = (_Float16)a3.z; a.f[15] = (_Float16)a3.w;

    sched_fence();   // all loads issued above; WMMAs below get partial waits

    #pragma unroll
    for (int m = 0; m < 3; ++m)
      #pragma unroll
      for (int nt = 0; nt < 4; ++nt)
        acc[m][nt] = wmma_f16(a.h, b[m * 4 + nt].h, acc[m][nt]);
  }

  // Epilogue. D layout: element i -> row (i + 8*g), col = nt*16 + ln.
  const long bidx = r0 / T_SEQ;
  const long t0   = (r0 % T_SEQ) + 8 * lg;

  #pragma unroll
  for (int nt = 0; nt < 4; ++nt) {
    const int h = nt * 16 + ln;
    #pragma unroll
    for (int i = 0; i < 8; ++i) {
      const long r = r0 + 8 * lg + i;
      qh[r * HDIM + h] = (_Float16)acc[0][nt][i];
      kh[r * HDIM + h] = (_Float16)acc[1][nt][i];
    }
    // v transposed: rows of the D fragment are contiguous in t -> one b128.
    union { _Float16 f[8]; u32x4 u; } pk;
    #pragma unroll
    for (int i = 0; i < 8; ++i) pk.f[i] = (_Float16)acc[2][nt][i];
    *(u32x4*)&vT[(bidx * HDIM + h) * T_SEQ + t0] = pk.u;
  }
}

// ---------------------------------------------------------------------------
// Kernel 2: flash attention with causal mask.
// Grid: (T/64, B). Block: 128 threads = 4 waves; wave w owns 16 query rows
// and runs fully independently (wave-private LDS region, no block barrier),
// so the causal trip-count imbalance between waves is free.
// ---------------------------------------------------------------------------
__global__ __launch_bounds__(128) void flash_head_kernel(
    const _Float16* __restrict__ qh,
    const _Float16* __restrict__ kh,
    const _Float16* __restrict__ vT,
    float* __restrict__ out)
{
  __shared__ __attribute__((aligned(16))) _Float16 ps[4][16 * 64]; // per-wave P

  const int tid  = threadIdx.x;
  const int wave = tid >> 5;
  const int lane = tid & 31;
  const int lg   = lane >> 4;
  const int ln   = lane & 15;

  const int qtile = blockIdx.x;        // 0..63
  const int bb    = blockIdx.y;        // 0..3

  const _Float16* Q = qh + (long)bb * T_SEQ * HDIM;
  const _Float16* K = kh + (long)bb * T_SEQ * HDIM;
  const _Float16* V = vT + (long)bb * HDIM * T_SEQ;   // [h][t]
  float* O          = out + (long)bb * T_SEQ * HDIM;

  const int qb = qtile * 64 + wave * 16;   // first query row of this wave

  // Q fragments (A layout), kept in registers for the whole kernel.
  Frag qa[2];
  {
    const _Float16* p = Q + (long)(qb + ln) * HDIM;
    #pragma unroll
    for (int hs = 0; hs < 2; ++hs) {
      qa[hs].u[0] = *(const u32x4*)(p + hs * 32 + 8 * lg);
      qa[hs].u[1] = *(const u32x4*)(p + hs * 32 + 16 + 8 * lg);
    }
  }

  v8f z = vzero();
  v8f accO[4];
  #pragma unroll
  for (int nt = 0; nt < 4; ++nt) accO[nt] = z;
  float mrow[8], lrow[8];
  #pragma unroll
  for (int i = 0; i < 8; ++i) { mrow[i] = -3.0e38f; lrow[i] = 0.0f; }

  const int qri = qb + 8 * lg;         // row of accumulator element i is qri+i
  _Float16* pw = &ps[wave][0];

  for (int kbase = 0; kbase <= qtile * 64; kbase += 64) {
    // ---- batch-load all 8 K fragments for this 64-key block ----
    Frag bk[4][2];
    #pragma unroll
    for (int kt = 0; kt < 4; ++kt) {
      const _Float16* kp = K + (long)(kbase + kt * 16 + ln) * HDIM;
      #pragma unroll
      for (int hs = 0; hs < 2; ++hs) {
        bk[kt][hs].u[0] = *(const u32x4*)(kp + hs * 32 + 8 * lg);
        bk[kt][hs].u[1] = *(const u32x4*)(kp + hs * 32 + 16 + 8 * lg);
      }
    }
    sched_fence();

    // ---- S = Q * K^T (8 WMMAs, partial waits) ----
    v8f S[4];
    #pragma unroll
    for (int kt = 0; kt < 4; ++kt) {
      v8f s = z;
      s = wmma_f16(qa[0].h, bk[kt][0].h, s);
      s = wmma_f16(qa[1].h, bk[kt][1].h, s);
      S[kt] = s;
    }

    // ---- scale + causal mask (diagonal 64-key block only) ----
    const bool diag = (kbase == qtile * 64);
    #pragma unroll
    for (int kt = 0; kt < 4; ++kt) {
      const int key = kbase + kt * 16 + ln;
      #pragma unroll
      for (int i = 0; i < 8; ++i) {
        float v = S[kt][i] * 0.125f;               // 1/sqrt(64)
        if (diag && key > qri + i) v = -3.0e38f;
        S[kt][i] = v;
      }
    }

    // ---- online softmax: row max, rescale, exp, row sum ----
    float scale[8];
    #pragma unroll
    for (int i = 0; i < 8; ++i) {
      float mx = fmaxf(fmaxf(S[0][i], S[1][i]), fmaxf(S[2][i], S[3][i]));
      #pragma unroll
      for (int d = 1; d < 16; d <<= 1)
        mx = fmaxf(mx, __shfl_xor(mx, d, 32));     // within 16-lane half-group
      float mn = fmaxf(mrow[i], mx);
      scale[i] = __expf(mrow[i] - mn);
      mrow[i]  = mn;
    }
    #pragma unroll
    for (int kt = 0; kt < 4; ++kt)
      #pragma unroll
      for (int i = 0; i < 8; ++i)
        S[kt][i] = __expf(S[kt][i] - mrow[i]);
    #pragma unroll
    for (int i = 0; i < 8; ++i) {
      float rs = (S[0][i] + S[1][i]) + (S[2][i] + S[3][i]);
      #pragma unroll
      for (int d = 1; d < 16; d <<= 1)
        rs += __shfl_xor(rs, d, 32);
      lrow[i] = lrow[i] * scale[i] + rs;
    }
    #pragma unroll
    for (int nt = 0; nt < 4; ++nt)
      #pragma unroll
      for (int i = 0; i < 8; ++i)
        accO[nt][i] *= scale[i];

    // ---- C-layout -> A-layout via wave-private LDS (f16) ----
    #pragma unroll
    for (int kt = 0; kt < 4; ++kt)
      #pragma unroll
      for (int i = 0; i < 8; ++i)
        pw[(8 * lg + i) * 64 + kt * 16 + ln] = (_Float16)S[kt][i];

    asm volatile("s_wait_dscnt 0x0" ::: "memory");  // same-wave DS RAW fence

    // ---- O += P * V (8 WMMAs); V fragments contiguous from vT[h][t] ----
    #pragma unroll
    for (int c = 0; c < 2; ++c) {
      Frag pa;
      const _Float16* pr = pw + ln * 64 + c * 32;
      pa.u[0] = *(const u32x4*)(pr + 8 * lg);
      pa.u[1] = *(const u32x4*)(pr + 16 + 8 * lg);
      Frag bv[4];                       // batch the 4 V-fragment loads
      #pragma unroll
      for (int nt = 0; nt < 4; ++nt) {
        const _Float16* vp =
            V + (long)(nt * 16 + ln) * T_SEQ + kbase + c * 32;
        bv[nt].u[0] = *(const u32x4*)(vp + 8 * lg);
        bv[nt].u[1] = *(const u32x4*)(vp + 16 + 8 * lg);
      }
      sched_fence();
      #pragma unroll
      for (int nt = 0; nt < 4; ++nt)
        accO[nt] = wmma_f16(pa.h, bv[nt].h, accO[nt]);
    }
  }

  // ---- epilogue: out = accO / l ----
  #pragma unroll
  for (int nt = 0; nt < 4; ++nt) {
    const int h = nt * 16 + ln;
    #pragma unroll
    for (int i = 0; i < 8; ++i)
      O[(long)(qri + i) * HDIM + h] = accO[nt][i] / lrow[i];
  }
}

// ---------------------------------------------------------------------------
extern "C" void kernel_launch(void* const* d_in, const int* in_sizes, int n_in,
                              void* d_out, int out_size, void* d_ws, size_t ws_size,
                              hipStream_t stream) {
  (void)in_sizes; (void)n_in; (void)out_size; (void)ws_size;
  const float* x  = (const float*)d_in[0];
  const float* Wk = (const float*)d_in[1];
  const float* Wq = (const float*)d_in[2];
  const float* Wv = (const float*)d_in[3];
  float* out = (float*)d_out;

  const long BT = 4L * T_SEQ;                  // 16384 rows
  _Float16* qh = (_Float16*)d_ws;              // [BT][64] f16
  _Float16* kh = qh + BT * HDIM;               // [BT][64] f16
  _Float16* vT = kh + BT * HDIM;               // [B][64][T] f16 (transposed)
  _Float16* wT = vT + BT * HDIM;               // [3][64][1024] f16

  w_transpose_kernel<<<(3 * C_EMB * HDIM) / 256, 256, 0, stream>>>(Wq, Wk, Wv, wT);
  qkv_proj_kernel<<<(int)(BT / 64), 128, 0, stream>>>(x, wT, qh, kh, vT);
  flash_head_kernel<<<dim3(T_SEQ / 64, 4), 128, 0, stream>>>(qh, kh, vT, out);
}
